// MultiHeadedAttentionFast_18416819765966
// MI455X (gfx1250) — matrix operand order repaired and checked by reading, not compiled, not run on hardware
//
#include <hip/hip_runtime.h>

// MI455X / gfx1250: wave32, WMMA f32_16x16x32_f16 (K=32 == head depth).
// Flash-style attention, two-pass softmax, f16 tensor-core matmuls.
// Attention processes TWO 16-query tiles per wave so K/V fragment loads are
// amortized over 2x the WMMAs and independent softmax VALU work fills the
// WMMA->VALU hazard slots.

typedef __attribute__((ext_vector_type(16))) _Float16 v16h;
typedef __attribute__((ext_vector_type(8)))  _Float16 v8h;
typedef __attribute__((ext_vector_type(8)))  float    v8f;

#define N_SP    4096   // 64*64 spatial tokens
#define IN_C    64
#define VAL_C   32
#define N_HEADS 4
#define BATCH   2
#define NBH     (BATCH * N_HEADS)
#define D_TOT   (N_HEADS * VAL_C)   // 128

__device__ __forceinline__ v16h cat8(v8h lo, v8h hi) {
  return __builtin_shufflevector(lo, hi, 0,1,2,3,4,5,6,7,8,9,10,11,12,13,14,15);
}

__device__ __forceinline__ v8f wmma_f16(v16h a, v16h b, v8f c) {
  // D = A(16x32 f16) * B(32x16 f16) + C(16x16 f32)
  return __builtin_amdgcn_wmma_f32_16x16x32_f16(
      /*neg_a=*/false, a, /*neg_b=*/false, b,
      /*c_mod=*/(short)0, c, /*reuse_a=*/false, /*reuse_b=*/false);
}

// ---------------------------------------------------------------------------
// Kernel 1: fused Q/K/V projection + f32->f16 conversion.
//   Qh[bh][n][d]  (token-major rows of 32 -> A-fragment source), pre-scaled 1/sqrt(32)
//   Kh[bh][n][d]  (token-major rows of 32 -> B-fragment source, lane = key column)
//   Vh[bh][d][n]  (depth-major -> B-fragment source, lane = depth column)
// ---------------------------------------------------------------------------
__global__ __launch_bounds__(256) void proj_kernel(
    const float* __restrict__ x,  const float* __restrict__ wq,
    const float* __restrict__ wk, const float* __restrict__ wv,
    _Float16* __restrict__ qh, _Float16* __restrict__ kh, _Float16* __restrict__ vh)
{
  const int n  = blockIdx.x * 256 + threadIdx.x;
  const int d  = blockIdx.y;
  const int bh = blockIdx.z;
  const int b = bh >> 2, h = bh & 3;
  const float* xb = x + (size_t)b * IN_C * N_SP + n;
  const float* q = wq + (h * VAL_C + d) * IN_C;   // uniform -> scalar loads
  const float* k = wk + (h * VAL_C + d) * IN_C;
  const float* v = wv + (h * VAL_C + d) * IN_C;
  float aq = 0.f, ak = 0.f, av = 0.f;
#pragma unroll
  for (int c = 0; c < IN_C; ++c) {
    float xv = xb[(size_t)c * N_SP];
    aq = fmaf(q[c], xv, aq);
    ak = fmaf(k[c], xv, ak);
    av = fmaf(v[c], xv, av);
  }
  const float inv_sqrt_d = 0.17677669529663689f;  // 1/sqrt(32) folded into Q
  qh[((size_t)bh * N_SP + n) * VAL_C + d] = (_Float16)(aq * inv_sqrt_d);
  kh[((size_t)bh * N_SP + n) * VAL_C + d] = (_Float16)ak;
  vh[((size_t)bh * VAL_C + d) * N_SP + n] = (_Float16)av;
}

// ---------------------------------------------------------------------------
// Kernel 2: attention. One wave32 per (bh, 32-query super-tile = 2x16 tiles).
// Pass 1: row max of S via WMMA; Pass 2: P = exp(S-max) -> LDS layout swap ->
// WMMA against V. Divide by (rowsum * 4096) at the end.
// ---------------------------------------------------------------------------
__global__ __launch_bounds__(32) void attn_kernel(
    const _Float16* __restrict__ qh, const _Float16* __restrict__ kh,
    const _Float16* __restrict__ vh, _Float16* __restrict__ attnh)
{
  __shared__ __align__(16) _Float16 Psh[2 * 16 * 32];  // two 16q x 32k P tiles

  const int lane = threadIdx.x;
  const int low  = lane & 15;
  const int hi   = lane >> 4;
  const int k0   = hi ? 8 : 0;      // 16-bit A-fragment half-K swizzle
  const int qsup = blockIdx.x;      // 32-query super-tile
  const int bh   = blockIdx.y;
  const int b = bh >> 2, h = bh & 3;

  const _Float16* Q = qh + (size_t)bh * N_SP * VAL_C;
  const _Float16* K = kh + (size_t)bh * VAL_C * N_SP;
  const _Float16* V = vh + (size_t)bh * N_SP * VAL_C;

  const int qbase = qsup * 32;
  // A fragments of Q^T (16 queries x 32 depth each). ISA 16-bit A 16x32 layout:
  // lanes 0-15: M=lane, K halves {0..7, 16..23}; lanes 16-31: M=lane-16, {8..15, 24..31}.
  const _Float16* qrow0 = Q + (size_t)(qbase + low) * VAL_C;
  const _Float16* qrow1 = Q + (size_t)(qbase + 16 + low) * VAL_C;
  const v16h a_q0 = cat8(*(const v8h*)(qrow0 + k0), *(const v8h*)(qrow0 + 16 + k0));
  const v16h a_q1 = cat8(*(const v8h*)(qrow1 + k0), *(const v8h*)(qrow1 + 16 + k0));

  const _Float16* krow = K + (size_t)low * VAL_C;   // B fragment: lane = key column

  // ---- pass 1: global row max of S for both query tiles ----
  float rmax0[8], rmax1[8];
#pragma unroll
  for (int r = 0; r < 8; ++r) { rmax0[r] = -3.0e38f; rmax1[r] = -3.0e38f; }
#pragma unroll 2
  for (int kb = 0; kb < N_SP; kb += 32) {
    v16h b0 = cat8(*(const v8h*)(krow + (size_t)kb * VAL_C + k0), *(const v8h*)(krow + (size_t)kb * VAL_C + 16 + k0));
    v16h b1 = cat8(*(const v8h*)(krow + (size_t)(kb + 16) * VAL_C + k0), *(const v8h*)(krow + (size_t)(kb + 16) * VAL_C + 16 + k0));
    v8f z = {0.f,0.f,0.f,0.f,0.f,0.f,0.f,0.f};
    v8f s00 = wmma_f16(a_q0, b0, z);
    v8f s01 = wmma_f16(a_q0, b1, z);
    v8f s10 = wmma_f16(a_q1, b0, z);
    v8f s11 = wmma_f16(a_q1, b1, z);
#pragma unroll
    for (int r = 0; r < 8; ++r) {
      rmax0[r] = fmaxf(rmax0[r], fmaxf(s00[r], s01[r]));
      rmax1[r] = fmaxf(rmax1[r], fmaxf(s10[r], s11[r]));
    }
  }
#pragma unroll
  for (int r = 0; r < 8; ++r) {                 // reduce across the 16-lane N group
    rmax0[r] = fmaxf(rmax0[r], __shfl_xor(rmax0[r], 1, 32));
    rmax0[r] = fmaxf(rmax0[r], __shfl_xor(rmax0[r], 2, 32));
    rmax0[r] = fmaxf(rmax0[r], __shfl_xor(rmax0[r], 4, 32));
    rmax0[r] = fmaxf(rmax0[r], __shfl_xor(rmax0[r], 8, 32));
    rmax1[r] = fmaxf(rmax1[r], __shfl_xor(rmax1[r], 1, 32));
    rmax1[r] = fmaxf(rmax1[r], __shfl_xor(rmax1[r], 2, 32));
    rmax1[r] = fmaxf(rmax1[r], __shfl_xor(rmax1[r], 4, 32));
    rmax1[r] = fmaxf(rmax1[r], __shfl_xor(rmax1[r], 8, 32));
  }

  // ---- pass 2: P = exp(S - max), O += P * V^T ----
  float lsum0[8], lsum1[8];
#pragma unroll
  for (int r = 0; r < 8; ++r) { lsum0[r] = 0.f; lsum1[r] = 0.f; }
  v8f o00 = {0.f,0.f,0.f,0.f,0.f,0.f,0.f,0.f};
  v8f o01 = {0.f,0.f,0.f,0.f,0.f,0.f,0.f,0.f};
  v8f o10 = {0.f,0.f,0.f,0.f,0.f,0.f,0.f,0.f};
  v8f o11 = {0.f,0.f,0.f,0.f,0.f,0.f,0.f,0.f};

  for (int kb = 0; kb < N_SP; kb += 32) {
    v16h b0 = cat8(*(const v8h*)(krow + (size_t)kb * VAL_C + k0), *(const v8h*)(krow + (size_t)kb * VAL_C + 16 + k0));
    v16h b1 = cat8(*(const v8h*)(krow + (size_t)(kb + 16) * VAL_C + k0), *(const v8h*)(krow + (size_t)(kb + 16) * VAL_C + 16 + k0));
    v8f z = {0.f,0.f,0.f,0.f,0.f,0.f,0.f,0.f};
    v8f s00 = wmma_f16(a_q0, b0, z);
    v8f s01 = wmma_f16(a_q0, b1, z);
    v8f s10 = wmma_f16(a_q1, b0, z);
    v8f s11 = wmma_f16(a_q1, b1, z);
    const int row = 8 * hi;                     // C-layout: VGPR r, lane group -> M
#pragma unroll
    for (int r = 0; r < 8; ++r) {
      float p00 = __expf(s00[r] - rmax0[r]);
      float p01 = __expf(s01[r] - rmax0[r]);
      float p10 = __expf(s10[r] - rmax1[r]);
      float p11 = __expf(s11[r] - rmax1[r]);
      lsum0[r] += p00 + p01;
      lsum1[r] += p10 + p11;
      Psh[(row + r) * 32 + low]            = (_Float16)p00;  // tile0, key = low
      Psh[(row + r) * 32 + 16 + low]       = (_Float16)p01;  // tile0, key = 16+low
      Psh[512 + (row + r) * 32 + low]      = (_Float16)p10;  // tile1
      Psh[512 + (row + r) * 32 + 16 + low] = (_Float16)p11;
    }
    // Single wave: drain our DS stores before the cross-lane A-fragment reload.
    asm volatile("s_wait_dscnt 0" ::: "memory");
    const _Float16* prow0 = Psh + low * 32;
    const _Float16* prow1 = Psh + 512 + low * 32;
    v16h a_p0 = cat8(*(const v8h*)(prow0 + k0), *(const v8h*)(prow0 + 16 + k0));
    v16h a_p1 = cat8(*(const v8h*)(prow1 + k0), *(const v8h*)(prow1 + 16 + k0));

    const _Float16* vrow0 = V + (size_t)low * N_SP + kb;  // B: lane = depth column
    const _Float16* vrow1 = V + (size_t)(16 + low) * N_SP + kb;
    v16h bv0 = cat8(*(const v8h*)(vrow0 + k0), *(const v8h*)(vrow0 + 16 + k0));
    v16h bv1 = cat8(*(const v8h*)(vrow1 + k0), *(const v8h*)(vrow1 + 16 + k0));
    o00 = wmma_f16(a_p0, bv0, o00);
    o01 = wmma_f16(a_p0, bv1, o01);
    o10 = wmma_f16(a_p1, bv0, o10);
    o11 = wmma_f16(a_p1, bv1, o11);
  }

#pragma unroll
  for (int r = 0; r < 8; ++r) {
    lsum0[r] += __shfl_xor(lsum0[r], 1, 32);
    lsum0[r] += __shfl_xor(lsum0[r], 2, 32);
    lsum0[r] += __shfl_xor(lsum0[r], 4, 32);
    lsum0[r] += __shfl_xor(lsum0[r], 8, 32);
    lsum1[r] += __shfl_xor(lsum1[r], 1, 32);
    lsum1[r] += __shfl_xor(lsum1[r], 2, 32);
    lsum1[r] += __shfl_xor(lsum1[r], 4, 32);
    lsum1[r] += __shfl_xor(lsum1[r], 8, 32);
  }

  // attnh[b][token][h*32 + d], f16, token-major for decode's B fragments.
  _Float16* ao = attnh + (size_t)b * D_TOT * N_SP;
#pragma unroll
  for (int r = 0; r < 8; ++r) {
    const float inv0 = 1.0f / (lsum0[r] * (float)N_SP);  // softmax norm + /(H*W)
    const float inv1 = 1.0f / (lsum1[r] * (float)N_SP);
    const int q0 = qbase + r + 8 * hi;
    const int q1 = qbase + 16 + r + 8 * hi;
    ao[(size_t)q0 * D_TOT + h * VAL_C + low]      = (_Float16)(o00[r] * inv0);
    ao[(size_t)q0 * D_TOT + h * VAL_C + 16 + low] = (_Float16)(o01[r] * inv0);
    ao[(size_t)q1 * D_TOT + h * VAL_C + low]      = (_Float16)(o10[r] * inv1);
    ao[(size_t)q1 * D_TOT + h * VAL_C + 16 + low] = (_Float16)(o11[r] * inv1);
  }
}

// ---------------------------------------------------------------------------
// Kernel 3: decode 1x1 conv: out(64 x 4096) = W(64x128) * attn(128x4096), fp32 out.
// One wave per 16x16 output tile, 4 chained WMMA over K=128.
// ---------------------------------------------------------------------------
__global__ __launch_bounds__(32) void decode_kernel(
    const float* __restrict__ w, const _Float16* __restrict__ attnh,
    float* __restrict__ out)
{
  const int lane = threadIdx.x, low = lane & 15, hi = lane >> 4;
  const int k0 = hi ? 8 : 0;
  const int nb = blockIdx.x * 16;
  const int mt = blockIdx.y;               // 0..3 (rows of 16 out channels)
  const int b  = blockIdx.z;
  const int m  = mt * 16 + low;

  v8f acc = {0.f,0.f,0.f,0.f,0.f,0.f,0.f,0.f};
  const _Float16* ab = attnh + (size_t)b * D_TOT * N_SP;
#pragma unroll
  for (int chunk = 0; chunk < 4; ++chunk) {
    const float* wr = w + (size_t)m * D_TOT + chunk * 32;
    v16h aw;                               // fp32 -> f16 A-fragment in registers
#pragma unroll
    for (int i = 0; i < 8; ++i) {
      aw[i]     = (_Float16)wr[k0 + i];
      aw[8 + i] = (_Float16)wr[16 + k0 + i];
    }
    const _Float16* br = ab + (size_t)(nb + low) * D_TOT + chunk * 32;
    v16h bt = cat8(*(const v8h*)(br + k0), *(const v8h*)(br + 16 + k0));
    acc = wmma_f16(aw, bt, acc);
  }
  float* ob = out + (size_t)b * 64 * N_SP;
#pragma unroll
  for (int r = 0; r < 8; ++r)
    ob[(size_t)(mt * 16 + r + 8 * hi) * N_SP + nb + low] = acc[r];
}

// ---------------------------------------------------------------------------
extern "C" void kernel_launch(void* const* d_in, const int* in_sizes, int n_in,
                              void* d_out, int out_size, void* d_ws, size_t ws_size,
                              hipStream_t stream) {
  (void)in_sizes; (void)n_in; (void)out_size; (void)ws_size;
  const float* x  = (const float*)d_in[0];
  const float* wq = (const float*)d_in[1];
  const float* wk = (const float*)d_in[2];
  const float* wv = (const float*)d_in[3];
  const float* wd = (const float*)d_in[4];
  float* out = (float*)d_out;

  char* ws = (char*)d_ws;
  const size_t seg = (size_t)NBH * N_SP * VAL_C * sizeof(_Float16); // 2 MiB
  _Float16* qh    = (_Float16*)(ws);
  _Float16* kh    = (_Float16*)(ws + seg);
  _Float16* vh    = (_Float16*)(ws + 2 * seg);
  _Float16* attnh = (_Float16*)(ws + 3 * seg);   // 8 MiB total

  proj_kernel  <<<dim3(N_SP / 256, VAL_C, NBH), 256, 0, stream>>>(x, wq, wk, wv, qh, kh, vh);
  attn_kernel  <<<dim3(N_SP / 32, NBH),          32, 0, stream>>>(qh, kh, vh, attnh);
  decode_kernel<<<dim3(N_SP / 16, 4, BATCH),     32, 0, stream>>>(wd, attnh, out);
}
